// TopKastLinear_84988812853626
// MI455X (gfx1250) — matrix operand order repaired and checked by reading, compiled, not run on hardware
//
#include <hip/hip_runtime.h>
#include <math.h>
#include <stdint.h>

typedef __bf16 v16bf __attribute__((ext_vector_type(16)));
typedef float  v8f   __attribute__((ext_vector_type(8)));
typedef unsigned int   u32;
typedef unsigned short u16;
typedef long long      s64;

// ---------------------------------------------------------------------------
// Workspace layout (bytes):
//   [0)            u32 histHi[65536]
//   [256K)         u32 histLo0[65536]
//   [512K)         u32 histLo1[65536]
//   [768K)         u32 sel[4] = {hiBin(k0), rankIn(k0), hiBin(k1), rankIn(k1)}
//   [768K+256)     float thresh
//   [1M)           bf16 A  (N*K)
//   [1M + 2*N*K)   bf16 Wmasked (OUT*K)
// ---------------------------------------------------------------------------

__device__ __forceinline__ u16 f2bf(float f) {
  u32 u = __float_as_uint(f);
  u += 0x7fffu + ((u >> 16) & 1u);   // round-to-nearest-even
  return (u16)(u >> 16);
}

// ------------------------- quantile selection ------------------------------

__global__ void tk_zero_u32(u32* p, int n) {
  int i = blockIdx.x * blockDim.x + threadIdx.x;
  if (i < n) p[i] = 0u;
}

__global__ void tk_hist_hi(const float* __restrict__ w, s64 n, u32* __restrict__ hist) {
  s64 i = (s64)blockIdx.x * blockDim.x + threadIdx.x;
  s64 stride = (s64)gridDim.x * blockDim.x;
  for (; i < n; i += stride) {
    u32 u = __float_as_uint(w[i]) & 0x7fffffffu;
    atomicAdd(&hist[u >> 16], 1u);
  }
}

__global__ void tk_find_bin(const u32* __restrict__ hist, u32* __restrict__ sel,
                            s64 k0, s64 k1) {
  __shared__ s64 seg[256];
  int t = threadIdx.x;
  s64 s = 0;
  for (int j = 0; j < 256; ++j) s += hist[t * 256 + j];
  seg[t] = s;
  __syncthreads();
  if (t == 0) {
    s64 ks[2] = {k0, k1};
    for (int which = 0; which < 2; ++which) {
      s64 k = ks[which];
      s64 cum = 0;
      int bin = 65535; s64 rin = 0;
      for (int sg = 0; sg < 256; ++sg) {
        if (cum + seg[sg] > k) {
          for (int j = 0; j < 256; ++j) {
            s64 c = (s64)hist[sg * 256 + j];
            if (cum + c > k) { bin = sg * 256 + j; rin = k - cum; goto done; }
            cum += c;
          }
        }
        cum += seg[sg];
      }
    done:
      sel[which * 2 + 0] = (u32)bin;
      sel[which * 2 + 1] = (u32)rin;
    }
  }
}

__global__ void tk_hist_lo(const float* __restrict__ w, s64 n, const u32* __restrict__ sel,
                           u32* __restrict__ lo0, u32* __restrict__ lo1) {
  u32 hi0 = sel[0], hi1 = sel[2];
  s64 i = (s64)blockIdx.x * blockDim.x + threadIdx.x;
  s64 stride = (s64)gridDim.x * blockDim.x;
  for (; i < n; i += stride) {
    u32 u = __float_as_uint(w[i]) & 0x7fffffffu;
    u32 h = u >> 16;
    if (h == hi0) atomicAdd(&lo0[u & 0xffffu], 1u);
    if (h == hi1) atomicAdd(&lo1[u & 0xffffu], 1u);
  }
}

__global__ void tk_find_val(const u32* __restrict__ lo0, const u32* __restrict__ lo1,
                            const u32* __restrict__ sel, float frac,
                            float* __restrict__ thresh_out) {
  __shared__ s64 seg[2][256];
  int t = threadIdx.x;
  {
    s64 s0 = 0, s1 = 0;
    for (int j = 0; j < 256; ++j) { s0 += lo0[t * 256 + j]; s1 += lo1[t * 256 + j]; }
    seg[0][t] = s0; seg[1][t] = s1;
  }
  __syncthreads();
  if (t == 0) {
    float v[2];
    for (int which = 0; which < 2; ++which) {
      const u32* h = which ? lo1 : lo0;
      s64 k = (s64)sel[which * 2 + 1];
      s64 cum = 0;
      int bin = 65535;
      for (int sg = 0; sg < 256; ++sg) {
        if (cum + seg[which][sg] > k) {
          for (int j = 0; j < 256; ++j) {
            s64 c = (s64)h[sg * 256 + j];
            if (cum + c > k) { bin = sg * 256 + j; goto found; }
            cum += c;
          }
        }
        cum += seg[which][sg];
      }
    found:
      v[which] = __uint_as_float((sel[which * 2 + 0] << 16) | (u32)bin);
    }
    *thresh_out = v[0] + frac * (v[1] - v[0]);
  }
}

// ------------------------- bf16 packing ------------------------------------

__global__ void tk_pack_a(const float* __restrict__ a, u16* __restrict__ out, s64 n) {
  s64 i = ((s64)blockIdx.x * blockDim.x + threadIdx.x) * 4;
  if (i >= n) return;
  float4 f = *(const float4*)(a + i);
  ushort4 r;
  r.x = f2bf(f.x); r.y = f2bf(f.y); r.z = f2bf(f.z); r.w = f2bf(f.w);
  *(ushort4*)(out + i) = r;
}

__global__ void tk_pack_w(const float* __restrict__ w, u16* __restrict__ out, s64 n,
                          const float* __restrict__ thresh_p) {
  s64 i = ((s64)blockIdx.x * blockDim.x + threadIdx.x) * 4;
  if (i >= n) return;
  float th = *thresh_p;
  float4 f = *(const float4*)(w + i);
  ushort4 r;
  r.x = (fabsf(f.x) >= th) ? f2bf(f.x) : (u16)0;
  r.y = (fabsf(f.y) >= th) ? f2bf(f.y) : (u16)0;
  r.z = (fabsf(f.z) >= th) ? f2bf(f.z) : (u16)0;
  r.w = (fabsf(f.w) >= th) ? f2bf(f.w) : (u16)0;
  *(ushort4*)(out + i) = r;
}

// ------------------------- WMMA GEMM ---------------------------------------
// out[n][o] = sum_k A[n][k] * W[o][k] + bias[o]
// Block tile 128x128, BK=32, double-buffered LDS filled by the gfx1250
// async-to-LDS engine (ASYNCcnt), consumed by v_wmma_f32_16x16x32_bf16.

#define BM 128
#define BN 128
#define BK2 32
#define LDPA 40   // padded LDS row stride (bf16 elems) = 80B: conflict-free frags

// one per-lane 16B global -> LDS async DMA (tracked by ASYNCcnt)
__device__ __forceinline__ void async_b128(u32 lds_off, const u16* g) {
  asm volatile("global_load_async_to_lds_b128 %0, %1, off"
               :: "v"(lds_off), "v"(g) : "memory");
}

__device__ __forceinline__ void wait_async0() {
#if defined(__has_builtin)
#if __has_builtin(__builtin_amdgcn_s_wait_asynccnt)
  __builtin_amdgcn_s_wait_asynccnt(0);
#else
  asm volatile("s_wait_asynccnt 0x0" ::: "memory");
#endif
#else
  asm volatile("s_wait_asynccnt 0x0" ::: "memory");
#endif
}

__global__ __launch_bounds__(256, 2) void tk_gemm(
    const u16* __restrict__ A, const u16* __restrict__ W,
    const float* __restrict__ bias, float* __restrict__ out,
    int N, int K, int OUT) {
  __shared__ __attribute__((aligned(16))) u16 As[2][BM][LDPA];
  __shared__ __attribute__((aligned(16))) u16 Bs[2][BN][LDPA];

  const int tid = threadIdx.x;
  const int lane = tid & 31;
  const int wid = tid >> 5;
  const int wm = wid >> 2;        // 0..1 -> 64 rows each
  const int wn = wid & 3;         // 0..3 -> 32 cols each
  const int lhalf = lane >> 4;    // 0/1
  const int l16 = lane & 15;

  const int blockM = blockIdx.y * BM;
  const int blockN = blockIdx.x * BN;

  // Global->LDS map: thread t owns 16 bf16 (2 x b128) of one row per phase.
  const int ldRow = tid >> 1;          // 0..127
  const int ldK = (tid & 1) * 16;      // 0 or 16

  const u16* aSrc = A + (size_t)(blockM + ldRow) * K + ldK;
  const u16* wSrc = W + (size_t)(blockN + ldRow) * K + ldK;

  const u32 aL[2] = {(u32)(uintptr_t)&As[0][ldRow][ldK],
                     (u32)(uintptr_t)&As[1][ldRow][ldK]};
  const u32 bL[2] = {(u32)(uintptr_t)&Bs[0][ldRow][ldK],
                     (u32)(uintptr_t)&Bs[1][ldRow][ldK]};

  v8f acc[4][2];
  {
    v8f z = {0.f, 0.f, 0.f, 0.f, 0.f, 0.f, 0.f, 0.f};
#pragma unroll
    for (int mi = 0; mi < 4; ++mi)
#pragma unroll
      for (int ni = 0; ni < 2; ++ni) acc[mi][ni] = z;
  }

  auto load_tile = [&](int buf, int kpos) {
    async_b128(aL[buf],      aSrc + kpos);
    async_b128(aL[buf] + 16, aSrc + kpos + 8);
    async_b128(bL[buf],      wSrc + kpos);
    async_b128(bL[buf] + 16, wSrc + kpos + 8);
  };

  auto compute_phase = [&](int b) {
    union FragU { v16bf v; uint4 q[2]; } fa[4], fb[2];
    // A 16x32 bf16 frag: lane<16 -> K 0..7 & 16..23 ; lane>=16 -> 8..15 & 24..31
#pragma unroll
    for (int mi = 0; mi < 4; ++mi) {
      const u16* p = &As[b][wm * 64 + mi * 16 + l16][lhalf * 8];
      fa[mi].q[0] = *(const uint4*)p;
      fa[mi].q[1] = *(const uint4*)(p + 16);
    }
    // B 32x16 bf16 frag: lane holds one column (= W row), lane<16 -> K 0..7 & 16..23 ; lane>=16 -> 8..15 & 24..31
#pragma unroll
    for (int ni = 0; ni < 2; ++ni) {
      const u16* p = &Bs[b][wn * 32 + ni * 16 + l16][lhalf * 8];
      fb[ni].q[0] = *(const uint4*)p;
      fb[ni].q[1] = *(const uint4*)(p + 16);
    }
#pragma unroll
    for (int mi = 0; mi < 4; ++mi)
#pragma unroll
      for (int ni = 0; ni < 2; ++ni)
        acc[mi][ni] = __builtin_amdgcn_wmma_f32_16x16x32_bf16(
            false, fa[mi].v, false, fb[ni].v, (short)0, acc[mi][ni], false, false);
  };

  // prologue: fill buffer 0
  load_tile(0, 0);

  for (int kb = 0; kb < K; kb += 2 * BK2) {
    // ---- phase 0: compute buf0, prefetch buf1 ----
    wait_async0();        // own async batch for buf0 landed
    __syncthreads();      // everyone's landed; previous compute on buf1 done
    load_tile(1, kb + BK2);
    compute_phase(0);
    // ---- phase 1: compute buf1, prefetch buf0 ----
    wait_async0();
    __syncthreads();
    if (kb + 2 * BK2 < K) load_tile(0, kb + 2 * BK2);
    compute_phase(1);
  }

  // Epilogue: C/D layout -> lane<16: M = r ; lane>=16: M = 8+r (VGPR r), N = l16
#pragma unroll
  for (int ni = 0; ni < 2; ++ni) {
    const int col = blockN + wn * 32 + ni * 16 + l16;
    const float bv = bias[col];
#pragma unroll
    for (int mi = 0; mi < 4; ++mi) {
      const int rowb = blockM + wm * 64 + mi * 16 + lhalf * 8;
#pragma unroll
      for (int r = 0; r < 8; ++r)
        out[(size_t)(rowb + r) * OUT + col] = acc[mi][ni][r] + bv;
    }
  }
}

// ---------------- TDM availability probe (never launched) ------------------
#if defined(__has_builtin)
#if __has_builtin(__builtin_amdgcn_tensor_load_to_lds) && __has_builtin(__builtin_amdgcn_s_wait_tensorcnt)
typedef unsigned int tdm_u32x4 __attribute__((ext_vector_type(4)));
typedef int tdm_i32x4 __attribute__((ext_vector_type(4)));
typedef int tdm_i32x8 __attribute__((ext_vector_type(8)));
__global__ void tk_tdm_probe() {
  tdm_u32x4 g0 = {0u, 0u, 0u, 0u};
  tdm_i32x8 g1 = {0, 0, 0, 0, 0, 0, 0, 0};
  tdm_i32x4 g2 = {0, 0, 0, 0};
  tdm_i32x4 g3 = {0, 0, 0, 0};
#if __clang_major__ >= 23
  tdm_i32x8 g4 = {0, 0, 0, 0, 0, 0, 0, 0};
  __builtin_amdgcn_tensor_load_to_lds(g0, g1, g2, g3, g4, 0);
#else
  __builtin_amdgcn_tensor_load_to_lds(g0, g1, g2, g3, 0);
#endif
  __builtin_amdgcn_s_wait_tensorcnt(0);
}
#endif
#endif

// ---------------------------------------------------------------------------

extern "C" void kernel_launch(void* const* d_in, const int* in_sizes, int n_in,
                              void* d_out, int out_size, void* d_ws, size_t ws_size,
                              hipStream_t stream) {
  const float* inputs = (const float*)d_in[0];
  const float* weight = (const float*)d_in[1];
  const float* bias   = (const float*)d_in[2];
  float* out = (float*)d_out;

  const s64 nA = (s64)in_sizes[0];
  const s64 nW = (s64)in_sizes[1];
  const int OUT = in_sizes[2];
  const int K = (int)(nW / OUT);
  const int N = (int)(nA / K);

  unsigned char* ws = (unsigned char*)d_ws;
  u32* histHi  = (u32*)ws;
  u32* histLo0 = (u32*)(ws + (size_t)65536 * 4);
  u32* histLo1 = (u32*)(ws + (size_t)65536 * 8);
  u32* sel     = (u32*)(ws + (size_t)65536 * 12);
  float* thr   = (float*)(ws + (size_t)65536 * 12 + 256);
  u16* aBf     = (u16*)(ws + ((size_t)1 << 20));
  u16* wBf     = (u16*)(ws + ((size_t)1 << 20) + (size_t)nA * 2);

  // jnp.quantile(., 0.9) 'linear': pos = q*(n-1), interp between ranks k0,k0+1
  double pos = 0.9 * (double)(nW - 1);
  s64 k0 = (s64)pos;
  if (k0 > nW - 1) k0 = nW - 1;
  s64 k1 = (k0 + 1 <= nW - 1) ? (k0 + 1) : k0;
  float frac = (float)(pos - (double)k0);

  const int zeroN = 65536 * 3 + 128;
  tk_zero_u32<<<(zeroN + 255) / 256, 256, 0, stream>>>((u32*)ws, zeroN);
  tk_hist_hi<<<2048, 256, 0, stream>>>(weight, nW, histHi);
  tk_find_bin<<<1, 256, 0, stream>>>(histHi, sel, k0, k1);
  tk_hist_lo<<<2048, 256, 0, stream>>>(weight, nW, sel, histLo0, histLo1);
  tk_find_val<<<1, 256, 0, stream>>>(histLo0, histLo1, sel, frac, thr);
  tk_pack_w<<<(int)((nW / 4 + 255) / 256), 256, 0, stream>>>(weight, wBf, nW, thr);
  tk_pack_a<<<(int)((nA / 4 + 255) / 256), 256, 0, stream>>>(inputs, aBf, nA);

  dim3 grid(OUT / BN, N / BM);
  tk_gemm<<<grid, 256, 0, stream>>>(aBf, wBf, bias, out, N, K, OUT);
}
